// SSAdaptiveFusion_34162169872661
// MI455X (gfx1250) — matrix operand order repaired and checked by reading, not compiled, not run on hardware
//
#include <hip/hip_runtime.h>
#include <hip/hip_bf16.h>

typedef _Float16 v16h __attribute__((ext_vector_type(16)));
typedef _Float16 v8h  __attribute__((ext_vector_type(8)));
typedef _Float16 v2h  __attribute__((ext_vector_type(2)));
typedef float    v8f  __attribute__((ext_vector_type(8)));

#define WMMA_F16(acc, A, B)                                                     \
  acc = __builtin_amdgcn_wmma_f32_16x16x32_f16(false, (A), false, (B),          \
                                               (short)0, (acc), false, false)

static constexpr int   CB          = 2;                  // batch
static constexpr int   CC          = 64;                 // channels
static constexpr int   CS          = 31 * 128 * 128;     // voxels per (b,c) = 507904
static constexpr int   TILES_PER_B = CS / 16;            // 31744 (exact)
static constexpr int   NPAIRS      = CB * TILES_PER_B / 2; // 31744
static constexpr int   XY_STRIDE   = 136;  // halves per transposed-xy row (272B = 17*16, 16B-aligned)
static constexpr int   HT_STRIDE   = 72;   // halves per transposed-h row (144B, 16B-aligned)
static constexpr float LOG2E       = 1.4426950408889634f;

// tanh(v) = 1 - 2/(exp2(2*log2e*v)+1)   -> mul, v_exp_f32, add, v_rcp_f32, fma
__device__ __forceinline__ float fast_tanh(float v) {
  float ex = __builtin_amdgcn_exp2f(v * (2.0f * LOG2E));
  float r  = __builtin_amdgcn_rcpf(1.0f + ex);   // inf -> 0, 0 -> 1 : saturates to +-1
  return 1.0f - 2.0f * r;
}
// sigmoid(z) = rcp(1 + exp2(-z*log2e))
__device__ __forceinline__ float fast_sigmoid(float z) {
  float ex = __builtin_amdgcn_exp2f(-z * LOG2E);
  return __builtin_amdgcn_rcpf(1.0f + ex);
}

__global__ __launch_bounds__(256)
void fused_gate_kernel(const float* __restrict__ x, const float* __restrict__ y,
                       const float* __restrict__ w1, const float* __restrict__ b1,
                       const float* __restrict__ w2, const float* __restrict__ b2,
                       float* __restrict__ out)
{
  // Inputs staged ONCE per tile as f16 hi/lo, transposed (voxel-major rows) so
  // every WMMA B-operand read is a contiguous ds_load_b128 with zero VALU.
  __shared__ __attribute__((aligned(16))) _Float16 s_xyhi[2][16][XY_STRIDE];
  __shared__ __attribute__((aligned(16))) _Float16 s_xylo[2][16][XY_STRIDE];
  __shared__ __attribute__((aligned(16))) _Float16 s_hhi [2][16][HT_STRIDE];
  __shared__ __attribute__((aligned(16))) _Float16 s_hlo [2][16][HT_STRIDE];

  const int tid  = threadIdx.x;
  const int lane = tid & 31;
  const int wave = tid >> 5;
  const int mrow = wave & 3;   // 16-output-channel M-tile owned by this wave
  const int g    = wave >> 2;  // group id (2 voxel tiles per block)
  const int gtid = tid & 127;
  const int Lm   = lane & 15;
  const int Lh   = lane >> 4;

  // -------- Weights resident in VGPRs in exact WMMA A-operand layout,
  // -------- split f16 hi + f16 lo (3-product GEMM ~ fp32 accuracy).
  v16h a1hi[4], a1lo[4], a2hi[2], a2lo[2];
  {
    const float* wr = w1 + (mrow * 16 + Lm) * 128;
    #pragma unroll
    for (int kc = 0; kc < 4; ++kc) {
      const int kb = kc * 32 + Lh * 8;
      #pragma unroll
      for (int i = 0; i < 8; ++i) {
        float f0 = wr[kb + i];        // halves 0..7 : K = kc*32 + Lh*8 + i
        float f1 = wr[kb + 16 + i];   // halves 8..15: K = kc*32 + 16 + Lh*8 + i
        _Float16 h0 = (_Float16)f0, h1 = (_Float16)f1;
        a1hi[kc][i]     = h0;  a1lo[kc][i]     = (_Float16)(f0 - (float)h0);
        a1hi[kc][i + 8] = h1;  a1lo[kc][i + 8] = (_Float16)(f1 - (float)h1);
      }
    }
    const float* wr2 = w2 + (mrow * 16 + Lm) * 64;
    #pragma unroll
    for (int kc = 0; kc < 2; ++kc) {
      const int kb = kc * 32 + Lh * 8;
      #pragma unroll
      for (int i = 0; i < 8; ++i) {
        float f0 = wr2[kb + i];
        float f1 = wr2[kb + 16 + i];
        _Float16 h0 = (_Float16)f0, h1 = (_Float16)f1;
        a2hi[kc][i]     = h0;  a2lo[kc][i]     = (_Float16)(f0 - (float)h0);
        a2hi[kc][i + 8] = h1;  a2lo[kc][i + 8] = (_Float16)(f1 - (float)h1);
      }
    }
  }
  float bias1[8], bias2[8];
  #pragma unroll
  for (int r = 0; r < 8; ++r) {   // C/D layout: acc[r] -> channel mrow*16 + Lh*8 + r
    bias1[r] = b1[mrow * 16 + Lh * 8 + r];
    bias2[r] = b2[mrow * 16 + Lh * 8 + r];
  }

  for (int p = blockIdx.x; p < NPAIRS; p += (int)gridDim.x) {
    const int t  = 2 * p + g;
    const int b  = (t >= TILES_PER_B) ? 1 : 0;
    const int s0 = (t - b * TILES_PER_B) * 16;
    const int planeBase = b * CC * CS + s0;

    __syncthreads();  // previous iteration's LDS consumers are done

    // -------- Stage: load 128ch x 16vox fp32 tile (coalesced 64B rows),
    // -------- split to f16 hi/lo ONCE, store transposed into LDS.
    #pragma unroll
    for (int j = 0; j < 8; ++j) {
      const int e     = gtid + j * 128;   // 0..1023
      const int cpair = e >> 4;           // channel pair 0..63 -> channels 2cp,2cp+1
      const int n     = e & 15;           // voxel within tile
      const int ch0   = 2 * cpair;
      const float* src = (ch0 < 64)
          ? (x + planeBase + ch0 * CS + n)
          : (y + planeBase + (ch0 - 64) * CS + n);
      float f0 = src[0];
      float f1 = src[CS];
      _Float16 h0 = (_Float16)f0, h1 = (_Float16)f1;
      v2h hp, lp;
      hp[0] = h0;                          hp[1] = h1;
      lp[0] = (_Float16)(f0 - (float)h0);  lp[1] = (_Float16)(f1 - (float)h1);
      *(v2h*)&s_xyhi[g][n][ch0] = hp;
      *(v2h*)&s_xylo[g][n][ch0] = lp;
    }
    // Prefetch next grid-stride tile (one 64B channel-row per thread).
    {
      const int pn = p + (int)gridDim.x;
      if (pn < NPAIRS) {
        const int tn  = 2 * pn + g;
        const int bn  = (tn >= TILES_PER_B) ? 1 : 0;
        const int sn  = (tn - bn * TILES_PER_B) * 16;
        const int pb  = bn * CC * CS + sn;
        const int row = gtid;  // 0..127
        const float* pf = (row < 64) ? (x + pb + row * CS)
                                     : (y + pb + (row - 64) * CS);
        __builtin_prefetch(pf, 0, 1);
      }
    }
    __syncthreads();

    // -------- GEMM1: h = tanh(w1 * [x;y] + b1); B operands straight from LDS.
    v8f acc = {};
    #pragma unroll
    for (int kc = 0; kc < 4; ++kc) {
      const _Float16* ph = &s_xyhi[g][Lm][kc * 32 + Lh * 8];
      const _Float16* pl = &s_xylo[g][Lm][kc * 32 + Lh * 8];
      v16h bhi, blo;
      #pragma unroll
      for (int i = 0; i < 16; ++i) { bhi[i] = ph[i + (i & 8)]; blo[i] = pl[i + (i & 8)]; }
      WMMA_F16(acc, a1hi[kc], bhi);
      WMMA_F16(acc, a1hi[kc], blo);
      WMMA_F16(acc, a1lo[kc], bhi);
    }
    // tanh + write h transposed as f16 hi/lo (one 16B chunk per lane per array)
    {
      v8h hh, hl;
      #pragma unroll
      for (int r = 0; r < 8; ++r) {
        float hv = fast_tanh(acc[r] + bias1[r]);
        _Float16 hi = (_Float16)hv;
        hh[r] = hi;
        hl[r] = (_Float16)(hv - (float)hi);
      }
      *(v8h*)&s_hhi[g][Lm][mrow * 16 + Lh * 8] = hh;
      *(v8h*)&s_hlo[g][Lm][mrow * 16 + Lh * 8] = hl;
    }
    __syncthreads();

    // -------- GEMM2: gate = sigmoid(w2 * h + b2)
    v8f acc2 = {};
    #pragma unroll
    for (int kc = 0; kc < 2; ++kc) {
      const _Float16* ph = &s_hhi[g][Lm][kc * 32 + Lh * 8];
      const _Float16* pl = &s_hlo[g][Lm][kc * 32 + Lh * 8];
      v16h bh, bl;
      #pragma unroll
      for (int i = 0; i < 16; ++i) { bh[i] = ph[i + (i & 8)]; bl[i] = pl[i + (i & 8)]; }
      WMMA_F16(acc2, a2hi[kc], bh);
      WMMA_F16(acc2, a2hi[kc], bl);
      WMMA_F16(acc2, a2lo[kc], bh);
    }

    // -------- Gated blend: x,y reconstructed from hi+lo LDS (~2^-22 rel).
    {
      const int chb = mrow * 16 + Lh * 8;
      const _Float16* pxh = &s_xyhi[g][Lm][chb];
      const _Float16* pxl = &s_xylo[g][Lm][chb];
      const _Float16* pyh = pxh + 64;   // y occupies halves 64..127
      const _Float16* pyl = pxl + 64;
      #pragma unroll
      for (int r = 0; r < 8; ++r) {
        float gate = fast_sigmoid(acc2[r] + bias2[r]);
        float xv = (float)pxh[r] + (float)pxl[r];
        float yv = (float)pyh[r] + (float)pyl[r];
        out[planeBase + (chb + r) * CS + Lm] = fmaf(gate, yv - xv, xv);
      }
    }
  }
}

extern "C" void kernel_launch(void* const* d_in, const int* in_sizes, int n_in,
                              void* d_out, int out_size, void* d_ws, size_t ws_size,
                              hipStream_t stream) {
  const float* x  = (const float*)d_in[0];
  const float* y  = (const float*)d_in[1];
  const float* w1 = (const float*)d_in[2];
  const float* b1 = (const float*)d_in[3];
  const float* w2 = (const float*)d_in[4];
  const float* b2 = (const float*)d_in[5];
  float* out = (float*)d_out;

  int grid = 4096;            // grid-stride: ~8 tile-pairs per block
  if (grid > NPAIRS) grid = NPAIRS;
  hipLaunchKernelGGL(fused_gate_kernel, dim3(grid), dim3(256), 0, stream,
                     x, y, w1, b1, w2, b2, out);
}
